// GraphSSMLayer_9672266350940
// MI455X (gfx1250) — compile-verified, matrix operand order checked by reading, not run on hardware
//
#include <hip/hip_runtime.h>
#include <math.h>

// ---------------------------------------------------------------------------
// Problem constants (GraphSSMLayer reference)
// ---------------------------------------------------------------------------
#define BB    16
#define HH    32
#define WW    32
#define CC    384
#define DI    768
#define LL    (HH*WW)          // 1024
#define MM    (BB*LL)          // 16384 tokens
#define HID   1536
#define DTRP  32               // dt_rank 24 / (24+2) = 26 padded to 32

// gfx1250 async global->LDS DMA (ASYNCcnt) if the toolchain exposes it
#if defined(__AMDGCN__) && __has_builtin(__builtin_amdgcn_global_load_async_to_lds_b128)
#define ASYNC_COPY 1
#else
#define ASYNC_COPY 0
#endif

#define AS1 __attribute__((address_space(1)))
#define AS3 __attribute__((address_space(3)))

typedef __bf16 bf16_t;
typedef __attribute__((ext_vector_type(16))) __bf16 v16bf;
typedef __attribute__((ext_vector_type(8)))  float  v8f;
typedef __attribute__((ext_vector_type(4)))  int    v4i;

union FragAB { v16bf v; unsigned u[8]; };
union FragC  { v8f   v; float    f[8]; };

__device__ __forceinline__ unsigned short f2bf(float f) {
  unsigned u = __float_as_uint(f);
  unsigned r = u + 0x7FFFu + ((u >> 16) & 1u);   // round-to-nearest-even
  return (unsigned short)(r >> 16);
}

__device__ __forceinline__ float siluf(float x) { return x / (1.0f + expf(-x)); }

__device__ __forceinline__ void async_b128(const void* g, void* l) {
#if ASYNC_COPY
  // signature (from clang diagnostic): (v4i AS1*, v4i AS3*, imm offset, imm cpol)
  __builtin_amdgcn_global_load_async_to_lds_b128(
      (AS1 v4i*)g, (AS3 v4i*)l, 0, 0);
#else
  *(uint4*)l = *(const uint4*)g;
#endif
}

__device__ __forceinline__ void wait_async() {
#if ASYNC_COPY
#if __has_builtin(__builtin_amdgcn_s_wait_asynccnt)
  __builtin_amdgcn_s_wait_asynccnt(0);
#else
  asm volatile("s_wait_asynccnt 0x0" ::: "memory");
#endif
#endif
}

// ---------------------------------------------------------------------------
// Wave32 WMMA GEMM, compile-time shapes: C[M,NN] = A[M,KK] * B[KK,NN]
//  - A: row-major bf16 activations (or f32 converted inline for AF32)
//  - B: weights pre-swizzled to packed-pair layout [KK/32][NN][16] uint32,
//       so one K-tile of B is a contiguous 8KB block.
//  - Double-buffered LDS; next tile copied global->LDS via async DMA
//    (GLOBAL_LOAD_ASYNC_TO_LDS_B128, ASYNCcnt) overlapping current WMMAs;
//    zero staging registers -> no scratch spills.
// MODE: 0 = f32 out
//       1 = +bias, softplus, f32 out            (dt projection)
//       2 = +residual, f32 out                  (w_out + skip)
//       3 = +bias, gelu(tanh), bf16 out         (fc1)
//       4 = +bias, +residual, f32 out           (fc2 + skip)
// ---------------------------------------------------------------------------
template<int BM, int BN, int KK, int NN, int WROWS, int WCOLS, int TM, int TN,
         int MODE, bool AF32>
__global__ __launch_bounds__(256, 1)
void gemm_wmma(const void* __restrict__ Aptr,
               const unsigned* __restrict__ Bpk,
               float* __restrict__ outF, unsigned short* __restrict__ outBf,
               const float* __restrict__ bias, const float* __restrict__ res)
{
  static_assert(BM == WROWS*TM*16 && BN == WCOLS*TN*16, "tile mismatch");
  static_assert(WROWS*WCOLS == 8, "8 waves");
  static_assert(KK % 32 == 0, "K multiple of 32");
  constexpr int LDT = 20;                        // uint stride: 16 K-pairs + 4 pad (16B)
  constexpr int KT  = KK / 32;                   // K tiles
  constexpr int A4  = BM * 4;                    // A uint4s per tile
  constexpr int B4  = BN * 4;                    // B uint4s per tile
  constexpr int AIT = (BM * 16) / 256;           // A scalar iters (f32 path)
  static_assert(AF32 || A4 % 256 == 0, "A staging");

  __shared__ alignas(16) unsigned ldsA[2][BM * LDT];
  __shared__ alignas(16) unsigned ldsB[2][BN * LDT];

  const int tid  = threadIdx.x;
  const int lane = tid & 31;
  const int wave = tid >> 5;
  const int wrow = wave % WROWS;
  const int wcol = wave / WROWS;
  const int row0 = blockIdx.y * BM;
  const int col0 = blockIdx.x * BN;
  const int ml   = lane & 15;                    // M (A frag) / N (B frag) index
  const int kg   = lane >> 4;                    // K-group select

  const unsigned short* Abf = (const unsigned short*)Aptr;
  const float*          Af  = (const float*)Aptr;

  FragC acc[TM][TN];
  #pragma unroll
  for (int i = 0; i < TM; ++i)
    #pragma unroll
    for (int j = 0; j < TN; ++j)
      #pragma unroll
      for (int r = 0; r < 8; ++r) acc[i][j].f[r] = 0.0f;

  // ---- copy one K-tile global -> LDS[buf]; async DMA when available ----
  auto copyTile = [&](int buf, int kt) {
    if constexpr (!AF32) {
      #pragma unroll
      for (int it = 0; it < A4 / 256; ++it) {
        int q = tid + it * 256;
        int m = q >> 2, c = q & 3;
        async_b128(Abf + (row0 + m) * KK + kt * 32 + c * 8,
                   &ldsA[buf][m * LDT + 4 * c]);
      }
    } else {
      #pragma unroll
      for (int it = 0; it < AIT; ++it) {         // inline f32 -> bf16 convert
        int i = tid + it * 256;
        int m = i >> 4, kp = i & 15;
        const float* ap = Af + (row0 + m) * KK + kt * 32 + 2 * kp;
        ldsA[buf][m * LDT + kp] =
            (unsigned)f2bf(ap[0]) | ((unsigned)f2bf(ap[1]) << 16);
      }
    }
    const unsigned* bp = Bpk + ((size_t)kt * NN + col0) * 16;
    if constexpr (B4 % 256 == 0) {
      #pragma unroll
      for (int it = 0; it < B4 / 256; ++it) {
        int q = tid + it * 256;
        int n = q >> 2, c = q & 3;
        async_b128(bp + q * 4, &ldsB[buf][n * LDT + 4 * c]);
      }
    } else {
      if (tid < B4) {
        int n = tid >> 2, c = tid & 3;
        async_b128(bp + tid * 4, &ldsB[buf][n * LDT + 4 * c]);
      }
    }
  };

  auto pipeSync = [&]() { wait_async(); __syncthreads(); };

  auto compute = [&](int buf) {
    FragAB afr[TM];
    #pragma unroll
    for (int i = 0; i < TM; ++i) {
      int mb = (wrow * TM + i) * 16 + ml;
      #pragma unroll
      for (int q = 0; q < 8; ++q) {
        int kp = (q >> 2) * 8 + kg * 4 + (q & 3);
        afr[i].u[q] = ldsA[buf][mb * LDT + kp];
      }
    }
    #pragma unroll
    for (int j = 0; j < TN; ++j) {              // stream B fragments: low pressure
      FragAB bfr;
      int nb = (wcol * TN + j) * 16 + ml;
      #pragma unroll
      for (int q = 0; q < 8; ++q) {
        int kp = (q >> 2) * 8 + kg * 4 + (q & 3);
        bfr.u[q] = ldsB[buf][nb * LDT + kp];
      }
      #pragma unroll
      for (int i = 0; i < TM; ++i)
        acc[i][j].v = __builtin_amdgcn_wmma_f32_16x16x32_bf16(
            false, afr[i].v, false, bfr.v, (short)0, acc[i][j].v, false, false);
    }
  };

  // ---- pipeline: DMA of tile k+1 into other buffer overlaps WMMA of tile k
  copyTile(0, 0);
  pipeSync();
  int cur = 0;
  for (int kt = 0; kt < KT; ++kt) {
    if (kt + 2 < KT) {                       // L2 prefetch of tile after next
      int kf = kt + 2;
      if constexpr (AF32)
        __builtin_prefetch(Af + (row0 + (tid >> 4)) * KK + kf * 32, 0, 1);
      else
        __builtin_prefetch(Abf + (row0 + (tid >> 4)) * KK + kf * 32, 0, 1);
      __builtin_prefetch(Bpk + ((size_t)kf * NN + col0) * 16 + tid * ((BN * 16) / 256), 0, 1);
    }
    if (kt + 1 < KT) copyTile(cur ^ 1, kt + 1);  // other buffer: no reader race
    compute(cur);
    if (kt + 1 < KT) { pipeSync(); cur ^= 1; }
  }

  // ---- fused epilogue (NN constexpr -> immediate store offsets) ----
  #pragma unroll
  for (int i = 0; i < TM; ++i) {
    int rowBase = row0 + (wrow * TM + i) * 16 + 8 * kg;
    #pragma unroll
    for (int j = 0; j < TN; ++j) {
      int col = col0 + (wcol * TN + j) * 16 + ml;
      int base = rowBase * NN + col;
      float bv = 0.0f;
      if constexpr (MODE == 1 || MODE == 3 || MODE == 4) bv = bias[col];
      #pragma unroll
      for (int r = 0; r < 8; ++r) {
        int oi = base + r * NN;
        float v = acc[i][j].f[r];
        if constexpr (MODE == 0) {
          outF[oi] = v;
        } else if constexpr (MODE == 1) {
          v += bv;
          outF[oi] = (v > 20.0f) ? v : log1pf(expf(v));
        } else if constexpr (MODE == 2) {
          outF[oi] = v + res[oi];
        } else if constexpr (MODE == 3) {
          v += bv;
          float t = 0.7978845608028654f * (v + 0.044715f * v * v * v);
          outBf[oi] = f2bf(0.5f * v * (1.0f + tanhf(t)));
        } else {
          v += bv;
          outF[oi] = v + res[oi];
        }
      }
    }
  }
}

// ---------------------------------------------------------------------------
// LayerNorm over C per token -> bf16
// ---------------------------------------------------------------------------
template<int T>
__global__ void ln_bf16_kernel(const float* __restrict__ x, const float* __restrict__ g,
                               const float* __restrict__ b, unsigned short* __restrict__ out,
                               int C, float eps)
{
  __shared__ float s1[T], s2[T];
  int t = threadIdx.x;
  const float* row = x + (size_t)blockIdx.x * C;
  float sum = 0.0f, sq = 0.0f;
  for (int c = t; c < C; c += T) { float v = row[c]; sum += v; sq += v * v; }
  s1[t] = sum; s2[t] = sq; __syncthreads();
  for (int s = T / 2; s > 0; s >>= 1) {
    if (t < s) { s1[t] += s1[t + s]; s2[t] += s2[t + s]; }
    __syncthreads();
  }
  float mu = s1[0] / C;
  float rstd = rsqrtf(s2[0] / C - mu * mu + eps);
  unsigned short* orow = out + (size_t)blockIdx.x * C;
  for (int c = t; c < C; c += T)
    orow[c] = f2bf((row[c] - mu) * rstd * g[c] + b[c]);
}

// ---------------------------------------------------------------------------
// Out-norm (C=768, eps 1e-5) * silu(z) -> bf16   (z = xz cols 768..1535)
// ---------------------------------------------------------------------------
__global__ void outnorm_gate_kernel(const float* __restrict__ y, const float* __restrict__ xz,
                                    const float* __restrict__ g, const float* __restrict__ b,
                                    unsigned short* __restrict__ out)
{
  constexpr int T = 256;
  __shared__ float s1[T], s2[T];
  int t = threadIdx.x;
  const float* row  = y  + (size_t)blockIdx.x * DI;
  const float* zrow = xz + (size_t)blockIdx.x * (2 * DI) + DI;
  float sum = 0.0f, sq = 0.0f;
  for (int c = t; c < DI; c += T) { float v = row[c]; sum += v; sq += v * v; }
  s1[t] = sum; s2[t] = sq; __syncthreads();
  for (int s = T / 2; s > 0; s >>= 1) {
    if (t < s) { s1[t] += s1[t + s]; s2[t] += s2[t + s]; }
    __syncthreads();
  }
  float mu = s1[0] / DI;
  float rstd = rsqrtf(s2[0] / DI - mu * mu + 1e-5f);
  unsigned short* orow = out + (size_t)blockIdx.x * DI;
  for (int c = t; c < DI; c += T) {
    float v = (row[c] - mu) * rstd * g[c] + b[c];
    orow[c] = f2bf(v * siluf(zrow[c]));
  }
}

// ---------------------------------------------------------------------------
// Depthwise 3x3 SAME conv + SiLU; in: xm = xz[:, 0:768]; out u (B,L,DI) f32
// ---------------------------------------------------------------------------
__global__ void dwconv_silu_kernel(const float* __restrict__ xz, const float* __restrict__ w,
                                   float* __restrict__ u)
{
  int idx = blockIdx.x * blockDim.x + threadIdx.x;
  if (idx >= MM * DI) return;
  int token = idx / DI;
  int d     = idx - token * DI;
  int bb = token >> 10, hw = token & 1023;
  int hh = hw >> 5, ww = hw & 31;
  const float* wk = w + d * 9;
  float s = 0.0f;
  #pragma unroll
  for (int dh = -1; dh <= 1; ++dh) {
    int h2 = hh + dh; if ((unsigned)h2 >= (unsigned)HH) continue;
    #pragma unroll
    for (int dw = -1; dw <= 1; ++dw) {
      int w2 = ww + dw; if ((unsigned)w2 >= (unsigned)WW) continue;
      int tok2 = (bb << 10) + (h2 << 5) + w2;
      s += wk[(dh + 1) * 3 + (dw + 1)] * xz[(size_t)tok2 * (2 * DI) + d];
    }
  }
  u[(size_t)token * DI + d] = siluf(s);
}

// ---------------------------------------------------------------------------
// Selective scan (DS=1 -> scalar state per channel). One lane per (b,d).
// y overwrites u in place (u[tok] read strictly before y[tok] write).
// ---------------------------------------------------------------------------
__global__ void scan_kernel(float* __restrict__ uy, const float* __restrict__ dt,
                            const float* __restrict__ xdbl, const float* __restrict__ A_log,
                            const float* __restrict__ Dp)
{
  int d  = blockIdx.x * blockDim.x + threadIdx.x;   // 0..767
  int bb = blockIdx.y;                              // 0..15
  float A  = -expf(A_log[d]);
  float Dd = Dp[d];
  float h = 0.0f;
  size_t base = (size_t)bb * LL;
  for (int l = 0; l < LL; ++l) {
    size_t tok = base + l;
    float uu  = uy[tok * DI + d];
    float dtt = dt[tok * DI + d];
    float Bc  = xdbl[tok * DTRP + 24];
    float Cc  = xdbl[tok * DTRP + 25];
    h = expf(dtt * A) * h + dtt * Bc * uu;
    uy[tok * DI + d] = h * Cc + uu * Dd;
  }
}

// ---------------------------------------------------------------------------
// Weight prep: f32 (rows x cols) -> bf16-pair packed [dstK/32][dstN][16] u32
// dst[( (k>>5)*dstN + n )*16 + (k>>1)&15] = pack(src[k][n], src[k+1][n])
// Zero-pads rows->dstK and cols->dstN.
// ---------------------------------------------------------------------------
__global__ void convert_pack_kernel(const float* __restrict__ src, unsigned* __restrict__ dst,
                                    int rows, int cols, int dstK, int dstN)
{
  int idx = blockIdx.x * blockDim.x + threadIdx.x;
  int total = (dstK / 32) * dstN * 16;
  if (idx >= total) return;
  int kp = idx & 15;
  int t  = idx >> 4;
  int n  = t % dstN;
  int kt = t / dstN;
  int k  = kt * 32 + 2 * kp;
  float lo = (k     < rows && n < cols) ? src[(size_t)k * cols + n]       : 0.0f;
  float hi = (k + 1 < rows && n < cols) ? src[(size_t)(k + 1) * cols + n] : 0.0f;
  dst[idx] = (unsigned)f2bf(lo) | ((unsigned)f2bf(hi) << 16);
}

// x_dbl[:, 0:24] -> bf16 padded to 32 cols (dt projection input, row-major A)
__global__ void dtin_prep_kernel(const float* __restrict__ xdbl, unsigned short* __restrict__ dtin)
{
  int idx = blockIdx.x * blockDim.x + threadIdx.x;
  if (idx >= MM * DTRP) return;
  int c = idx & (DTRP - 1);
  dtin[idx] = (c < 24) ? f2bf(xdbl[idx]) : (unsigned short)0;
}

// ---------------------------------------------------------------------------
extern "C" void kernel_launch(void* const* d_in, const int* in_sizes, int n_in,
                              void* d_out, int out_size, void* d_ws, size_t ws_size,
                              hipStream_t stream) {
  (void)in_sizes; (void)n_in; (void)out_size; (void)ws_size;
  const float* x         = (const float*)d_in[0];
  const float* ln1_g     = (const float*)d_in[1];
  const float* ln1_b     = (const float*)d_in[2];
  const float* w_in      = (const float*)d_in[3];
  const float* conv_w    = (const float*)d_in[4];
  const float* x_proj_w  = (const float*)d_in[5];
  const float* dt_proj_w = (const float*)d_in[6];
  const float* dt_proj_b = (const float*)d_in[7];
  const float* A_log     = (const float*)d_in[8];
  const float* Dp        = (const float*)d_in[9];
  const float* on_g      = (const float*)d_in[10];
  const float* on_b      = (const float*)d_in[11];
  const float* w_out     = (const float*)d_in[12];
  const float* ln2_g     = (const float*)d_in[13];
  const float* ln2_b     = (const float*)d_in[14];
  const float* fc1_w     = (const float*)d_in[15];
  const float* fc1_b     = (const float*)d_in[16];
  const float* fc2_w     = (const float*)d_in[17];
  const float* fc2_b     = (const float*)d_in[18];
  float* out = (float*)d_out;

  // -------- workspace carve-up (256B aligned, aliased after liveness ends) --
  char* base = (char*)d_ws;
  size_t off = 0;
  auto carve = [&](size_t bytes) -> char* {
    char* p = base + off;
    off += (bytes + 255) & ~(size_t)255;
    return p;
  };
  unsigned*       wInPk  = (unsigned*)carve((size_t)CC * (2*DI) * 2);   // packed pair u32
  unsigned*       xprPk  = (unsigned*)carve((size_t)DI * DTRP * 2);
  unsigned*       dtwPk  = (unsigned*)carve((size_t)DTRP * DI * 2);
  unsigned*       wOutPk = (unsigned*)carve((size_t)DI * CC * 2);
  unsigned*       fc1Pk  = (unsigned*)carve((size_t)CC * HID * 2);
  unsigned*       fc2Pk  = (unsigned*)carve((size_t)HID * CC * 2);
  unsigned short* h1Bf   = (unsigned short*)carve((size_t)MM * CC * 2);
  float*          xzF    = (float*)carve((size_t)MM * (2*DI) * 4);   // later: mlp bf16
  float*          uF     = (float*)carve((size_t)MM * DI * 4);       // u -> y -> x_mid
  float*          xdblF  = (float*)carve((size_t)MM * DTRP * 4);
  unsigned short* dtInBf = (unsigned short*)carve((size_t)MM * DTRP * 2);
  float*          dtF    = (float*)carve((size_t)MM * DI * 4);       // later: ynorm | h2
  unsigned short* ynBf   = (unsigned short*)dtF;                     // after scan
  unsigned short* h2Bf   = (unsigned short*)((char*)dtF + (size_t)MM * DI * 2);
  unsigned short* mlpBf  = (unsigned short*)xzF;                     // after gate
  float*          xmidF  = uF;                                       // after scan/gate

  // -------- weight conversion to packed-pair bf16 layout --------
  auto pk = [&](const float* s, unsigned* d, int r, int c, int dK, int dN) {
    int tot = (dK / 32) * dN * 16;
    convert_pack_kernel<<<(tot + 255) / 256, 256, 0, stream>>>(s, d, r, c, dK, dN);
  };
  pk(w_in,      wInPk,  CC, 2*DI, CC, 2*DI);
  pk(x_proj_w,  xprPk,  DI, 26,   DI, DTRP);
  pk(dt_proj_w, dtwPk,  24, DI,   DTRP, DI);
  pk(w_out,     wOutPk, DI, CC,   DI, CC);
  pk(fc1_w,     fc1Pk,  CC, HID,  CC, HID);
  pk(fc2_w,     fc2Pk,  HID, CC,  HID, CC);

  // -------- LN1 -> h1 (bf16) --------
  ln_bf16_kernel<128><<<MM, 128, 0, stream>>>(x, ln1_g, ln1_b, h1Bf, CC, 1e-6f);

  // -------- GEMM1: h1 @ w_in -> xz (f32, M x 1536) --------
  gemm_wmma<128,128,CC,2*DI,4,2,2,4,0,false><<<dim3((2*DI)/128, MM/128), 256, 0, stream>>>(
      h1Bf, wInPk, xzF, nullptr, nullptr, nullptr);

  // -------- depthwise conv 3x3 + SiLU -> u (f32) --------
  dwconv_silu_kernel<<<(MM * DI) / 256, 256, 0, stream>>>(xzF, conv_w, uF);

  // -------- x_proj: u @ x_proj_w(pad 32) -> x_dbl (f32, M x 32) --------
  gemm_wmma<128,32,DI,DTRP,8,1,1,2,0,true><<<dim3(DTRP/32, MM/128), 256, 0, stream>>>(
      uF, xprPk, xdblF, nullptr, nullptr, nullptr);

  // -------- dt input prep (pad 24 -> 32, bf16) --------
  dtin_prep_kernel<<<(MM * DTRP) / 256, 256, 0, stream>>>(xdblF, dtInBf);

  // -------- dt_proj: dt_in @ dt_proj_w + b, softplus -> dt (f32, M x 768) --
  gemm_wmma<128,128,DTRP,DI,4,2,2,4,1,false><<<dim3(DI/128, MM/128), 256, 0, stream>>>(
      dtInBf, dtwPk, dtF, nullptr, dt_proj_b, nullptr);

  // -------- selective scan (y in place over u) --------
  scan_kernel<<<dim3(DI/256, BB), 256, 0, stream>>>(uF, dtF, xdblF, A_log, Dp);

  // -------- out-norm * silu(z) -> ynorm (bf16) --------
  outnorm_gate_kernel<<<MM, 256, 0, stream>>>(uF, xzF, on_g, on_b, ynBf);

  // -------- GEMM2: ynorm @ w_out + x -> x_mid (f32, M x 384) --------
  gemm_wmma<128,128,DI,CC,4,2,2,4,2,false><<<dim3(CC/128, MM/128), 256, 0, stream>>>(
      ynBf, wOutPk, xmidF, nullptr, nullptr, x);

  // -------- LN2 -> h2 (bf16) --------
  ln_bf16_kernel<128><<<MM, 128, 0, stream>>>(xmidF, ln2_g, ln2_b, h2Bf, CC, 1e-6f);

  // -------- GEMM3: h2 @ fc1 + b, gelu -> mlp (bf16, M x 1536) --------
  gemm_wmma<128,128,CC,HID,4,2,2,4,3,false><<<dim3(HID/128, MM/128), 256, 0, stream>>>(
      h2Bf, fc1Pk, nullptr, mlpBf, fc1_b, nullptr);

  // -------- GEMM4: mlp @ fc2 + b + x_mid -> out (f32, M x 384) --------
  gemm_wmma<128,128,HID,CC,4,2,2,4,4,false><<<dim3(CC/128, MM/128), 256, 0, stream>>>(
      mlpBf, fc2Pk, out, nullptr, fc2_b, xmidF);
}
